// BitNetLinear_73280732005188
// MI455X (gfx1250) — 8-bit matrix operand order repaired and checked by reading, compiled, not run on hardware
//
#include <hip/hip_runtime.h>

typedef int v8i __attribute__((ext_vector_type(8)));

#define EPSQ 1e-8f

#if __has_builtin(__builtin_amdgcn_global_load_async_to_lds_b128) && \
    __has_builtin(__builtin_amdgcn_s_wait_asynccnt)
#define USE_ASYNC_LDS 1
typedef int v4i_gcc __attribute__((vector_size(16)));
typedef __attribute__((address_space(1))) v4i_gcc gbl_v4i;
typedef __attribute__((address_space(3))) v4i_gcc lds_v4i;
#define TO_GBL(p) ((gbl_v4i*)(uintptr_t)(p))
#define TO_LDS(p) ((lds_v4i*)(unsigned)(uintptr_t)(p))
#else
#define USE_ASYNC_LDS 0
#endif


__global__ void init_ws_kernel(unsigned* wsu) {
    if (threadIdx.x < 16) wsu[threadIdx.x] = 0u;
}

__global__ void absmax_kernel(const float4* __restrict__ x, unsigned* wsu, int n4) {
    __shared__ float red[256];
    float m = 0.f;
    for (int i = blockIdx.x * blockDim.x + threadIdx.x; i < n4; i += gridDim.x * blockDim.x) {
        float4 v = x[i];
        m = fmaxf(m, fmaxf(fmaxf(fabsf(v.x), fabsf(v.y)), fmaxf(fabsf(v.z), fabsf(v.w))));
    }
    red[threadIdx.x] = m;
    __syncthreads();
    for (int s = 128; s > 0; s >>= 1) {
        if ((int)threadIdx.x < s) red[threadIdx.x] = fmaxf(red[threadIdx.x], red[threadIdx.x + s]);
        __syncthreads();
    }
    if (threadIdx.x == 0) atomicMax(wsu, __float_as_uint(red[0]));
}

__global__ void wabs_partial_kernel(const float4* __restrict__ w, float* partial, int n4) {
    __shared__ float red[256];
    float s = 0.f;
    for (int i = blockIdx.x * 256 + threadIdx.x; i < n4; i += 256 * 256) {
        float4 v = w[i];
        s += fabsf(v.x) + fabsf(v.y) + fabsf(v.z) + fabsf(v.w);
    }
    red[threadIdx.x] = s;
    __syncthreads();
    for (int st = 128; st > 0; st >>= 1) {
        if ((int)threadIdx.x < st) red[threadIdx.x] += red[threadIdx.x + st];
        __syncthreads();
    }
    if (threadIdx.x == 0) partial[blockIdx.x] = red[0];
}

__global__ void finalize_kernel(const float* __restrict__ bias, const float* __restrict__ partial,
                                unsigned* wsu, float* wsf, float* __restrict__ bq) {
    __shared__ float red[256];
    const int t = threadIdx.x;

    red[t] = partial[t];
    __syncthreads();
    for (int s = 128; s > 0; s >>= 1) {
        if (t < s) red[t] += red[t + s];
        __syncthreads();
    }
    const float gamma_w = red[0] * (1.0f / (1024.f * 1024.f));
    __syncthreads();

    float4 bv = ((const float4*)bias)[t];
    red[t] = fabsf(bv.x) + fabsf(bv.y) + fabsf(bv.z) + fabsf(bv.w);
    __syncthreads();
    for (int s = 128; s > 0; s >>= 1) {
        if (t < s) red[t] += red[t + s];
        __syncthreads();
    }
    const float gamma_b = red[0] * (1.0f / 1024.f);

    const float inv_b = 1.0f / (gamma_b + EPSQ);
    float4 q;
    q.x = fminf(fmaxf(rintf(bv.x * inv_b), -1.f), 1.f) * gamma_b;
    q.y = fminf(fmaxf(rintf(bv.y * inv_b), -1.f), 1.f) * gamma_b;
    q.z = fminf(fmaxf(rintf(bv.z * inv_b), -1.f), 1.f) * gamma_b;
    q.w = fminf(fmaxf(rintf(bv.w * inv_b), -1.f), 1.f) * gamma_b;
    ((float4*)bq)[t] = q;

    if (t == 0) {
        float maxv  = __uint_as_float(wsu[0]);
        float scale = exp2f(floorf(log2f(maxv * (1.0f / 127.0f) + EPSQ)));
        wsf[1] = gamma_w;
        wsf[2] = scale;
        wsf[3] = gamma_w * scale;
    }
}

__global__ void quantw_kernel(const float4* __restrict__ w, const float* __restrict__ wsf,
                              char4* __restrict__ wq, int n4) {
    int i = blockIdx.x * blockDim.x + threadIdx.x;
    if (i >= n4) return;
    const float inv = 1.0f / (wsf[1] + EPSQ);
    float4 v = w[i];
    char4 q;
    q.x = (signed char)__float2int_rn(fminf(fmaxf(v.x * inv, -1.f), 1.f));
    q.y = (signed char)__float2int_rn(fminf(fmaxf(v.y * inv, -1.f), 1.f));
    q.z = (signed char)__float2int_rn(fminf(fmaxf(v.z * inv, -1.f), 1.f));
    q.w = (signed char)__float2int_rn(fminf(fmaxf(v.w * inv, -1.f), 1.f));
    wq[i] = q;
}

__device__ __forceinline__ unsigned quant_pack4(float4 v, float inv_s) {
    int b0 = __float2int_rn(fminf(fmaxf(v.x * inv_s, -127.f), 127.f));
    int b1 = __float2int_rn(fminf(fmaxf(v.y * inv_s, -127.f), 127.f));
    int b2 = __float2int_rn(fminf(fmaxf(v.z * inv_s, -127.f), 127.f));
    int b3 = __float2int_rn(fminf(fmaxf(v.w * inv_s, -127.f), 127.f));
    return (unsigned)(b0 & 255) | ((unsigned)(b1 & 255) << 8) |
           ((unsigned)(b2 & 255) << 16) | ((unsigned)(b3 & 255) << 24);
}

union FragA { v8i v; unsigned long long u[4]; };
union FragB { v8i v; int4 q[2]; };

__global__ __launch_bounds__(256)
void gemm_iu8_kernel(const float* __restrict__ x, const signed char* __restrict__ wq8,
                     const float* __restrict__ bq, const float* __restrict__ wsf,
                     float* __restrict__ out) {
    __shared__ signed char As[2][128 * 64];
    __shared__ signed char Bs[2][128 * 64];

    const int tid  = threadIdx.x;
    const int lane = tid & 31;
    const int wv   = tid >> 5;
    const int n0   = blockIdx.x * 128;
    const int m0   = blockIdx.y * 128;

    const float inv_s = 1.0f / wsf[2];
    const float alpha = wsf[3];

    const int mw = (wv & 3) * 32;
    const int nw = (wv >> 2) * 64;
    const int lr = lane & 15;
    const int hA = (lane < 16) ? 0 : 16;
    const int hB = (lane < 16) ? 0 : 16;

    const int    arow = tid >> 4;
    const int    akb  = (tid & 15) * 4;
    const float* gA   = x + (size_t)(m0 + arow) * 1024 + akb;
    const int    aoff = arow * 64 + akb;

    const int    brow = tid >> 2;
    const int    bkb  = (tid & 3) * 16;
    const signed char* gB = wq8 + (size_t)(n0 + brow) * 1024 + bkb;
    const int    boff = brow * 64 + bkb;

    v8i acc[2][4];
    const v8i zero = {0, 0, 0, 0, 0, 0, 0, 0};
#pragma unroll
    for (int mi = 0; mi < 2; ++mi)
#pragma unroll
        for (int ni = 0; ni < 4; ++ni) acc[mi][ni] = zero;

    float4 areg[8];
#if !USE_ASYNC_LDS
    int4 breg[2];
#endif

#pragma unroll
    for (int it = 0; it < 8; ++it)
        areg[it] = *(const float4*)(gA + (size_t)(it * 16) * 1024);
#if USE_ASYNC_LDS
#pragma unroll
    for (int it = 0; it < 2; ++it)
        __builtin_amdgcn_global_load_async_to_lds_b128(
            TO_GBL(gB + (size_t)(it * 64) * 1024),
            TO_LDS(&Bs[0][boff + it * 4096]), 0, 0);
#else
#pragma unroll
    for (int it = 0; it < 2; ++it)
        breg[it] = *(const int4*)(gB + (size_t)(it * 64) * 1024);
#endif
#pragma unroll
    for (int it = 0; it < 8; ++it)
        *(unsigned*)&As[0][aoff + it * 1024] = quant_pack4(areg[it], inv_s);
#if USE_ASYNC_LDS
    __builtin_amdgcn_s_wait_asynccnt(0);
#else
#pragma unroll
    for (int it = 0; it < 2; ++it)
        *(int4*)&Bs[0][boff + it * 4096] = breg[it];
#endif
    __syncthreads();

    for (int kt = 0; kt < 16; ++kt) {
        const int cur = kt & 1;
        const int nxt = cur ^ 1;
        const int k1  = (kt + 1) * 64;

        if (kt < 15) {
#pragma unroll
            for (int it = 0; it < 8; ++it)
                areg[it] = *(const float4*)(gA + (size_t)(it * 16) * 1024 + k1);
#if USE_ASYNC_LDS
#pragma unroll
            for (int it = 0; it < 2; ++it)
                __builtin_amdgcn_global_load_async_to_lds_b128(
                    TO_GBL(gB + (size_t)(it * 64) * 1024 + k1),
                    TO_LDS(&Bs[nxt][boff + it * 4096]), 0, 0);
#else
#pragma unroll
            for (int it = 0; it < 2; ++it)
                breg[it] = *(const int4*)(gB + (size_t)(it * 64) * 1024 + k1);
#endif
            if (kt < 14) {
#pragma unroll
                for (int it = 0; it < 8; ++it)
                    __builtin_prefetch(gA + (size_t)(it * 16) * 1024 + k1 + 64, 0, 0);
            }
        }

        FragA a[2];
#pragma unroll
        for (int mi = 0; mi < 2; ++mi) {
            const signed char* p = &As[cur][(mw + mi * 16 + lr) * 64 + hA];
            a[mi].u[0] = *(const unsigned long long*)(p);
            a[mi].u[1] = *(const unsigned long long*)(p + 8);
            a[mi].u[2] = *(const unsigned long long*)(p + 32);
            a[mi].u[3] = *(const unsigned long long*)(p + 40);
        }
        FragB b[4];
#pragma unroll
        for (int ni = 0; ni < 4; ++ni) {
            const signed char* p = &Bs[cur][(nw + ni * 16 + lr) * 64 + hB];
            b[ni].q[0] = *(const int4*)(p);
            b[ni].q[1] = *(const int4*)(p + 32);
        }
#pragma unroll
        for (int ni = 0; ni < 4; ++ni)
#pragma unroll
            for (int mi = 0; mi < 2; ++mi)
                acc[mi][ni] = __builtin_amdgcn_wmma_i32_16x16x64_iu8(
                    true, a[mi].v, true, b[ni].v, acc[mi][ni], false, false);

        if (kt < 15) {
#pragma unroll
            for (int it = 0; it < 8; ++it)
                *(unsigned*)&As[nxt][aoff + it * 1024] = quant_pack4(areg[it], inv_s);
#if !USE_ASYNC_LDS
#pragma unroll
            for (int it = 0; it < 2; ++it)
                *(int4*)&Bs[nxt][boff + it * 4096] = breg[it];
#endif
        }
#if USE_ASYNC_LDS
        __builtin_amdgcn_s_wait_asynccnt(0);
#endif
        __syncthreads();
    }

#pragma unroll
    for (int ni = 0; ni < 4; ++ni) {
        int colg = n0 + nw + ni * 16 + lr;
        float bb = bq[colg];
#pragma unroll
        for (int mi = 0; mi < 2; ++mi) {
            int rbase = m0 + mw + mi * 16 + ((lane >> 4) << 3);
#pragma unroll
            for (int r = 0; r < 8; ++r) {
                out[(size_t)(rbase + r) * 1024 + colg] = alpha * (float)acc[mi][ni][r] + bb;
            }
        }
    }
}

extern "C" void kernel_launch(void* const* d_in, const int* in_sizes, int n_in,
                              void* d_out, int out_size, void* d_ws, size_t ws_size,
                              hipStream_t stream) {
    const float* x      = (const float*)d_in[0];
    const float* weight = (const float*)d_in[1];
    const float* bias   = (const float*)d_in[2];
    float* out = (float*)d_out;

    char* ws = (char*)d_ws;
    unsigned* wsu     = (unsigned*)ws;
    float*    wsf     = (float*)ws;
    float*    partial = (float*)(ws + 64);
    float*    bq      = (float*)(ws + 4096);
    signed char* wq8  = (signed char*)(ws + 8192);

    const int n_x  = in_sizes[0];
    const int n_w  = in_sizes[1];
    const int M    = n_x / 1024;

    hipLaunchKernelGGL(init_ws_kernel, dim3(1), dim3(64), 0, stream, wsu);
    hipLaunchKernelGGL(absmax_kernel, dim3(2048), dim3(256), 0, stream,
                       (const float4*)x, wsu, n_x / 4);
    hipLaunchKernelGGL(wabs_partial_kernel, dim3(256), dim3(256), 0, stream,
                       (const float4*)weight, partial, n_w / 4);
    hipLaunchKernelGGL(finalize_kernel, dim3(1), dim3(256), 0, stream,
                       bias, partial, wsu, wsf, bq);
    hipLaunchKernelGGL(quantw_kernel, dim3(1024), dim3(256), 0, stream,
                       (const float4*)weight, wsf, (char4*)wq8, n_w / 4);
    hipLaunchKernelGGL(gemm_iu8_kernel, dim3(8, M / 128), dim3(256), 0, stream,
                       x, wq8, bq, wsf, out);
}
